// fePAM_15204184228134
// MI455X (gfx1250) — matrix operand order repaired and checked by reading, not compiled, not run on hardware
//
#include <hip/hip_runtime.h>

// Problem constants (fixed by the reference).
#define B_ 2
#define C_ 64
#define H_ 128
#define W_ 256
#define K_ 32
#define N_ (H_*W_)   // 32768 pixels

typedef __attribute__((ext_vector_type(16))) _Float16 v16h;
typedef __attribute__((ext_vector_type(8)))  _Float16 v8h;
typedef __attribute__((ext_vector_type(8)))  float    v8f;
typedef __attribute__((ext_vector_type(4)))  unsigned int u32x4;
typedef __attribute__((ext_vector_type(8)))  int      i32x8;
typedef __attribute__((ext_vector_type(4)))  int      i32x4;

#define HAVE_TDM __has_builtin(__builtin_amdgcn_tensor_load_to_lds)

// ---------------------------------------------------------------------------
// Kernel 1: per-batch transpose + convert  [C, N] f32  ->  [N, C] f16.
// Makes every gathered key/value row one contiguous 128B block (one cache
// line) and halves the L2 gather traffic that bounds this workload.
// ---------------------------------------------------------------------------
__global__ void xpose_cvt_f16(const float* __restrict__ in, _Float16* __restrict__ out) {
  int b  = blockIdx.y;
  int n0 = blockIdx.x * 64;
  const float* inb = in  + (size_t)b * C_ * N_;
  _Float16*   outb = out + (size_t)b * N_ * C_;
  __shared__ _Float16 tile[64][72];
  int tn = threadIdx.x & 63, q4 = threadIdx.x >> 6;
#pragma unroll
  for (int i = 0; i < 16; ++i) {
    int c = i * 4 + q4;
    tile[tn][c] = (_Float16)inb[(size_t)c * N_ + n0 + tn];
  }
  __syncthreads();
#pragma unroll
  for (int i = 0; i < 16; ++i) {
    int nn = i * 4 + q4;
    outb[(size_t)(n0 + nn) * C_ + tn] = tile[nn][tn];
  }
}

#if HAVE_TDM
// Issue one TDM gather-mode descriptor: pull 16 rows (128B each, f16[64]) of
// `base` selected by the positions held in lanes [first_lane, first_lane+16)
// into LDS at `ldsoff`, with +16B pad per row (144B stride, matches vals[][72]).
static __device__ __forceinline__ void tdm_gather16(unsigned ldsoff,
                                                    const _Float16* base,
                                                    int pos_lane, int first_lane) {
  unsigned long long ga = (unsigned long long)(uintptr_t)base;
  u32x4 g0;
  g0[0] = 0x80000001u;                              // gather_mode=1, idx16, count=1
  g0[1] = ldsoff;                                   // lds_addr
  g0[2] = (unsigned)(ga & 0xffffffffu);             // global_addr[31:0]
  g0[3] = (unsigned)((ga >> 32) & 0x01ffffffu) | (2u << 30);  // addr[56:32] | type=2
  i32x8 g1;
  g1[0] = (1 << 16) | (1 << 20) | (4 << 22) | (3 << 25); // data_size=2B, pad_en,
                                                         // pad every 32 DW, +4 DW
  g1[1] = (64 << 16);                               // tensor_dim0 = 64 (channels)
  g1[2] = (int)(32768u << 16);                      // tensor_dim1 = N rows
  g1[3] = (64 << 16);                               // tile_dim0 = 64
  g1[4] = 16;                                       // tile_dim1 = #indices
  g1[5] = 64;                                       // tensor_dim0_stride = 64
  g1[6] = 0;
  g1[7] = 0;
  i32x4 g2, g3;
#pragma unroll
  for (int i = 0; i < 4; ++i) {
    g2[i] = __builtin_amdgcn_readlane(pos_lane, first_lane + 2 * i) |
            (__builtin_amdgcn_readlane(pos_lane, first_lane + 2 * i + 1) << 16);
    g3[i] = __builtin_amdgcn_readlane(pos_lane, first_lane + 8 + 2 * i) |
            (__builtin_amdgcn_readlane(pos_lane, first_lane + 8 + 2 * i + 1) << 16);
  }
#if defined(__clang_major__) && __clang_major__ >= 23
  i32x8 gx = {};
  __builtin_amdgcn_tensor_load_to_lds(g0, g1, g2, g3, gx, 0);
#else
  __builtin_amdgcn_tensor_load_to_lds(g0, g1, g2, g3, 0);
#endif
}
#endif

// ---------------------------------------------------------------------------
// Kernel 2: per-pixel attention. One wave32 per pixel, lane = candidate.
//   scores = Key[32x64].q    -> 2 cand-tiles x 2 ch-chunks WMMA (B = q bcast)
//   softmax over 32 lanes    -> shfl_xor reductions
//   out = attn.Val[32x64]    -> 4 ch-tiles WMMA (A = attn bcast, B from LDS)
// Value rows are gathered into LDS by the Tensor Data Mover (gather mode,
// 16 row-indices per descriptor), overlapping with the score WMMAs.
// ---------------------------------------------------------------------------
__global__ void attn32(const _Float16* __restrict__ Qt,
                       const _Float16* __restrict__ St,
                       const _Float16* __restrict__ Rt,
                       const int* __restrict__ Px,
                       const int* __restrict__ Py,
                       float* __restrict__ Ot,
                       int direct_out) {
  const int pix  = blockIdx.x;          // b*N + n
  const int b    = pix >> 15;           // / N_
  const int n    = pix & (N_ - 1);
  const int lane = threadIdx.x;         // 0..31

  __shared__ __align__(16) _Float16 vals[32][72];  // value rows, 144B stride
  __shared__ float scores[32];
  __shared__ __align__(16) _Float16 attn_h[32];

  // Candidate position owned by this lane.
  int px  = Px[(size_t)n * K_ + lane];
  int py  = Py[(size_t)n * K_ + lane];
  int pos = px * W_ + py;

  // ---- stage the 32 value rows (128B each) into LDS ----
#if HAVE_TDM
  {
    unsigned ldsbase = (unsigned)(uintptr_t)(&vals[0][0]);
    const _Float16* rbase = Rt + (size_t)b * N_ * C_;
    tdm_gather16(ldsbase,            rbase, pos, 0);   // candidates 0..15
    tdm_gather16(ldsbase + 16 * 144, rbase, pos, 16);  // candidates 16..31
  }
#else
  {
    const uint4* src = reinterpret_cast<const uint4*>(Rt + ((size_t)b * N_ + pos) * C_);
    uint4* dst = reinterpret_cast<uint4*>(&vals[lane][0]);
#pragma unroll
    for (int i = 0; i < 8; ++i) dst[i] = src[i];
  }
#endif

  const int kb16 = (lane < 16) ? 0 : 16;   // B-fragment K base
  const int kb8  = (lane < 16) ? 0 : 8;    // unused (A fragments also use kb16)

  // Replicated-query B fragments (all 16 columns identical), per channel chunk.
  const _Float16* qrow = Qt + ((size_t)b * N_ + n) * C_;
  v16h bq0 = *reinterpret_cast<const v16h*>(qrow + kb16);
  v16h bq1 = *reinterpret_cast<const v16h*>(qrow + 32 + kb16);

  // Scores: keys load straight from global into A fragments (contiguous 16B runs).
  v8f acc0 = {}; v8f acc1 = {};
#pragma unroll
  for (int t = 0; t < 2; ++t) {
    int m    = t * 16 + (lane & 15);
    int mpos = __shfl(pos, m, 32);
    const _Float16* krow = St + ((size_t)b * N_ + mpos) * C_;
#pragma unroll
    for (int c2 = 0; c2 < 2; ++c2) {
      v8h lo = *reinterpret_cast<const v8h*>(krow + c2 * 32 + kb16);
      v8h hi = *reinterpret_cast<const v8h*>(krow + c2 * 32 + kb16 + 8);
      v16h a;
#pragma unroll
      for (int i = 0; i < 8; ++i) { a[i] = lo[i]; a[i + 8] = hi[i]; }
      v16h bb = c2 ? bq1 : bq0;
      if (t == 0)
        acc0 = __builtin_amdgcn_wmma_f32_16x16x32_f16(false, a, false, bb, (short)0, acc0, false, false);
      else
        acc1 = __builtin_amdgcn_wmma_f32_16x16x32_f16(false, a, false, bb, (short)0, acc1, false, false);
    }
  }

  // Spill scores (all columns of D identical): lane0 covers M=0..7, lane16 M=8..15.
  if (lane == 0) {
#pragma unroll
    for (int i = 0; i < 8; ++i) { scores[i] = acc0[i]; scores[16 + i] = acc1[i]; }
  }
  if (lane == 16) {
#pragma unroll
    for (int i = 0; i < 8; ++i) { scores[8 + i] = acc0[i]; scores[24 + i] = acc1[i]; }
  }
  __syncthreads();

  // Softmax over the 32 candidates (lane k owns candidate k).
  float s = scores[lane];
  float mx = s;
#pragma unroll
  for (int d = 16; d > 0; d >>= 1) mx = fmaxf(mx, __shfl_xor(mx, d, 32));
  float e = __expf(s - mx);
  float sum = e;
#pragma unroll
  for (int d = 16; d > 0; d >>= 1) sum += __shfl_xor(sum, d, 32);
  attn_h[lane] = (_Float16)(e / sum);
  __syncthreads();

  // A fragment: attention weights replicated across all 16 rows.
  v16h aA;
  {
    v8h lo = *reinterpret_cast<const v8h*>(&attn_h[kb16]);
    v8h hi = *reinterpret_cast<const v8h*>(&attn_h[kb16 + 8]);
#pragma unroll
    for (int i = 0; i < 8; ++i) { aA[i] = lo[i]; aA[i + 8] = hi[i]; }
  }

#if HAVE_TDM
  __builtin_amdgcn_s_wait_tensorcnt(0);   // TDM writes to LDS complete
#endif

  float* orow = Ot + ((size_t)b * N_ + n) * C_;
#pragma unroll
  for (int t = 0; t < 4; ++t) {
    int ch = t * 16 + (lane & 15);
    v16h bv;                              // B = Val[32 cand x 16 ch tile], from LDS
#pragma unroll
    for (int i = 0; i < 16; ++i) bv[i] = vals[kb16 + i][ch];
    v8f d = {};
    d = __builtin_amdgcn_wmma_f32_16x16x32_f16(false, aA, false, bv, (short)0, d, false, false);
    if (lane < 16) {
      if (direct_out)
        Ot[((size_t)b * C_ + t * 16 + lane) * N_ + n] = d[0];  // fallback: strided
      else
        orow[t * 16 + lane] = d[0];                            // contiguous [B,N,C]
    }
  }
}

// ---------------------------------------------------------------------------
// Kernel 3: transpose back  [B, N, C] f32  ->  [B, C, N] f32 (both coalesced).
// ---------------------------------------------------------------------------
__global__ void xpose_out(const float* __restrict__ in, float* __restrict__ out) {
  int b  = blockIdx.y;
  int n0 = blockIdx.x * 64;
  const float* inb = in  + (size_t)b * N_ * C_;
  float*      outb = out + (size_t)b * C_ * N_;
  __shared__ float tile[64][65];
  int c = threadIdx.x & 63, q4 = threadIdx.x >> 6;
#pragma unroll
  for (int i = 0; i < 16; ++i) {
    int nn = i * 4 + q4;
    tile[nn][c] = inb[(size_t)(n0 + nn) * C_ + c];
  }
  __syncthreads();
#pragma unroll
  for (int i = 0; i < 16; ++i) {
    int cc = i * 4 + q4;
    outb[(size_t)cc * N_ + n0 + c] = tile[c][cc];
  }
}

extern "C" void kernel_launch(void* const* d_in, const int* in_sizes, int n_in,
                              void* d_out, int out_size, void* d_ws, size_t ws_size,
                              hipStream_t stream) {
  (void)in_sizes; (void)n_in; (void)out_size;
  const float* Q = (const float*)d_in[0];
  const float* S = (const float*)d_in[1];
  const float* R = (const float*)d_in[2];
  const int*  Px = (const int*)d_in[3];
  const int*  Py = (const int*)d_in[4];

  char* ws = (char*)d_ws;
  const size_t half_sz = (size_t)B_ * N_ * C_ * sizeof(_Float16);  // 8 MiB
  const size_t ot_sz   = (size_t)B_ * N_ * C_ * sizeof(float);     // 16 MiB
  _Float16* Qt = (_Float16*)(ws);
  _Float16* St = (_Float16*)(ws + half_sz);
  _Float16* Rt = (_Float16*)(ws + 2 * half_sz);
  int direct = (ws_size < 3 * half_sz + ot_sz) ? 1 : 0;            // skip Ot if tight
  float* Ot = direct ? (float*)d_out : (float*)(ws + 3 * half_sz);

  dim3 tgrid(N_ / 64, B_);
  xpose_cvt_f16<<<tgrid, 256, 0, stream>>>(Q, Qt);
  xpose_cvt_f16<<<tgrid, 256, 0, stream>>>(S, St);
  xpose_cvt_f16<<<tgrid, 256, 0, stream>>>(R, Rt);
  attn32<<<dim3(B_ * N_), 32, 0, stream>>>(Qt, St, Rt, Px, Py, Ot, direct);
  if (!direct)
    xpose_out<<<tgrid, 256, 0, stream>>>(Ot, (float*)d_out);
}
